// MPAttentionInflationBlock_39384850104942
// MI455X (gfx1250) — compile-verified, matrix operand order checked by reading, not run on hardware
//
#include <hip/hip_runtime.h>

typedef __attribute__((ext_vector_type(16))) _Float16 v16h;
typedef __attribute__((ext_vector_type(8)))  _Float16 v8h;
typedef __attribute__((ext_vector_type(8)))  float    v8f;

#define LIN_EPS 1e-4f
#define MP_T    0.3f

// LDS strides (elements)
#define QS 1544   // qkv f32 row stride (16 rows x 1536 cols)
#define SF 520    // res0 / ycur f32 row stride
#define SA 528    // a16 f16 row stride (multiple of 8 -> 16B aligned frags)

// workspace layout (element offsets)
#define WS_WQKV_H   0                    // 2*1536*512 f16
#define WS_WOUT_H   (2*1536*512)         // 2*512*512 f16
#define WS_WPROJ_H  (2*1536*512 + 2*512*512)
#define WS_HALFS    (2*1536*512 + 2*512*512 + 512*512)
#define WS_MKV_BYTE ((size_t)WS_HALFS * 2)   // then 2*4096 f32 (pixel-normed mem_kv)

__device__ __forceinline__ float wave_red_add(float v) {
  #pragma unroll
  for (int o = 16; o; o >>= 1) v += __shfl_xor(v, o, 32);
  return v;
}

// Load one 16x32 f16 fragment (A layout; B uses the symmetric layout) from a
// row-major [row][K] array: halfs {k..k+7} and {k+16..k+23} per lane.
__device__ __forceinline__ v16h frag_ld(const _Float16* p) {
  v8h lo = *(const v8h*)p;
  v8h hi = *(const v8h*)(p + 16);
  return __builtin_shufflevector(lo, hi, 0,1,2,3,4,5,6,7,8,9,10,11,12,13,14,15);
}

// CDNA5 async global->LDS copy (ASYNCcnt path); lds_off is the wave-relative
// LDS byte offset (low 32 bits of the generic shared pointer).
__device__ __forceinline__ void async_ld_b128(unsigned lds_off, const void* g) {
  asm volatile("global_load_async_to_lds_b128 %0, %1, off"
               :: "v"(lds_off), "v"(g) : "memory");
}
__device__ __forceinline__ void wait_async0() {
  asm volatile("s_wait_asynccnt 0x0" ::: "memory");
}

// ---- prep kernel 1: l2-normalize weight rows -> f16 (gain folded into proj)
__global__ __launch_bounds__(256) void norm_weights(
    const float* __restrict__ wqkv, const float* __restrict__ wout,
    const float* __restrict__ wproj, const float* __restrict__ gain,
    _Float16* __restrict__ ws16) {
  int row  = blockIdx.x * 8 + (threadIdx.x >> 5);  // 4608 rows total
  int lane = threadIdx.x & 31;
  const float* src; _Float16* dst; float extra = 1.f;
  if (row < 3072)      { src = wqkv  + (size_t)row * 512;        dst = ws16 + WS_WQKV_H  + (size_t)row * 512; }
  else if (row < 4096) { int r = row - 3072; src = wout  + (size_t)r * 512; dst = ws16 + WS_WOUT_H  + (size_t)r * 512; }
  else                 { int r = row - 4096; src = wproj + (size_t)r * 512; dst = ws16 + WS_WPROJ_H + (size_t)r * 512; extra = gain[0]; }
  float vals[16], ss = 0.f;
  #pragma unroll
  for (int i = 0; i < 16; ++i) { float v = src[lane * 16 + i]; vals[i] = v; ss += v * v; }
  ss = wave_red_add(ss);
  float s = extra * 0.04419417382415922f / fmaxf(sqrtf(ss), LIN_EPS); // 1/sqrt(512)
  #pragma unroll
  for (int i = 0; i < 16; ++i) dst[lane * 16 + i] = (_Float16)(vals[i] * s);
}

// ---- prep kernel 2: pixel-norm mem_kv rows (128 rows of 64) -> f32
__global__ __launch_bounds__(256) void norm_memkv(
    const float* __restrict__ memkv, float* __restrict__ mkvn) {
  int row  = blockIdx.x * 8 + (threadIdx.x >> 5);  // 0..127
  int lane = threadIdx.x & 31;
  const float* src = memkv + (size_t)row * 64;
  float a = src[lane * 2], b = src[lane * 2 + 1];
  float ss = wave_red_add(a * a + b * b);
  float s  = 8.f / fmaxf(sqrtf(ss), LIN_EPS);      // l2norm * sqrt(64)
  mkvn[row * 64 + lane * 2]     = a * s;
  mkvn[row * 64 + lane * 2 + 1] = b * s;
}

// ---- main kernel: one workgroup per (b,h,w) row; 8 waves
__global__ __launch_bounds__(256) void mp_attn_main(
    const float* __restrict__ x, const _Float16* __restrict__ ws16,
    const float* __restrict__ mkvn, float* __restrict__ out) {
  extern __shared__ char smem_raw[];
  float*    qkv  = (float*)smem_raw;            // 16 x QS f32
  float*    res0 = qkv  + 16 * QS;              // 16 x SF f32 (original y)
  float*    ycur = res0 + 16 * SF;              // 16 x SF f32 (current y)
  float*    mkv  = ycur + 16 * SF;              // [2][8][4][64] f32 (layer's mem kv)
  _Float16* a16  = (_Float16*)(mkv + 4096);     // 16 x SA f16 (WMMA A staging)

  const _Float16* WQKV  = ws16 + WS_WQKV_H;
  const _Float16* WOUT  = ws16 + WS_WOUT_H;
  const _Float16* WPROJ = ws16 + WS_WPROJ_H;

  const int r  = blockIdx.x;
  const int bb = r >> 10, hw = r & 1023, hh = hw >> 5, ww = hw & 31;
  const size_t xbase = (size_t)bb * 8388608 + hh * 32 + ww; // b-stride 512*16*1024
  const int tid = threadIdx.x, lane = tid & 31, wv = tid >> 5;
  const int mrow = lane & 15;            // fragment row within tile
  const int kOff = (lane >> 4) << 3;     // 0 or 8
  const int mb   = (lane >> 4) << 3;     // accumulator M base
  const float rs = 1.3130643285972254f;  // 1/sqrt((1-t)^2 + t^2), t=0.3
  const unsigned mkv_lds = (unsigned)(size_t)mkv;  // wave-relative LDS byte offset

  // gather y tile [16 tokens x 512 ch]; c-stride = 16384, t-stride = 1024
  for (int idx = tid; idx < 8192; idx += 256) {
    int cc = idx >> 4, tt = idx & 15;
    float v = x[xbase + (size_t)cc * 16384 + (size_t)tt * 1024];
    res0[tt * SF + cc] = v;
    ycur[tt * SF + cc] = v;
    a16 [tt * SA + cc] = (_Float16)v;
  }

  for (int l = 0; l < 2; ++l) {
    // async-stage this layer's pixel-normed mem KV (16 KB) straight into LDS
    for (int idx = tid; idx < 1024; idx += 256)
      async_ld_b128(mkv_lds + idx * 16, (const char*)(mkvn + l * 4096) + idx * 16);
    wait_async0();
    __syncthreads();

    const _Float16* WQ = WQKV + (size_t)l * 1536 * 512;
    const _Float16* WO = WOUT + (size_t)l * 512 * 512;

    // GEMM1: qkv[16x1536] = y[16x512] @ wn_qkv^T   (96 N-tiles, 12 per wave)
    for (int j = 0; j < 12; ++j) {
      int n0 = (wv * 12 + j) * 16;
      if (j + 1 < 12)
        __builtin_prefetch(WQ + (size_t)(n0 + 16 + mrow) * 512 + kOff, 0, 0);
      v8f acc = {};
      #pragma unroll 4
      for (int kb = 0; kb < 16; ++kb) {
        int k0 = kb * 32 + kOff;
        v16h a = frag_ld(a16 + mrow * SA + k0);
        v16h b = frag_ld(WQ + (size_t)(n0 + mrow) * 512 + k0);
        acc = __builtin_amdgcn_wmma_f32_16x16x32_f16(false, a, false, b,
                                                     (short)0, acc, false, false);
      }
      #pragma unroll
      for (int i = 0; i < 8; ++i) qkv[(mb + i) * QS + n0 + mrow] = acc[i];
    }
    __syncthreads();

    // pixel-norm q,k,v (in place); wave 'wv' owns head h=wv
    const int h = wv;
    for (int row = 0; row < 48; ++row) {
      int part = row >> 4, tt = row & 15;
      float* p = qkv + tt * QS + part * 512 + h * 64;
      float a = p[lane * 2], b = p[lane * 2 + 1];
      float ss = wave_red_add(a * a + b * b);
      float nrm = fmaxf(sqrtf(ss), LIN_EPS);
      float s = (part == 0) ? (1.f / nrm)   // pixel_norm(q) * DIM_HEAD^-0.5
                            : (8.f / nrm);  // pixel_norm(k|v)
      p[lane * 2] = a * s; p[lane * 2 + 1] = b * s;
    }
    // attention: lane pair (m, m+16) splits D=64 in half; all 32 lanes active
    {
      int m = lane & 15;
      int dbase = (lane >> 4) * 32;
      const float* qrow = qkv + m * QS + h * 64 + dbase;
      float p[20], mx = -1e30f;
      for (int j = 0; j < 20; ++j) {
        const float* krow = ((j < 4) ? (mkv + h * 256 + j * 64)
                                     : (qkv + (j - 4) * QS + 512 + h * 64)) + dbase;
        float s = 0.f;
        #pragma unroll 8
        for (int d = 0; d < 32; ++d) s += qrow[d] * krow[d];
        s += __shfl_xor(s, 16, 32);       // combine the two half-dots
        p[j] = s; mx = fmaxf(mx, s);
      }
      float sum = 0.f;
      #pragma unroll
      for (int j = 0; j < 20; ++j) { p[j] = __expf(p[j] - mx); sum += p[j]; }
      float inv = 1.f / sum;
      float o[32];
      #pragma unroll
      for (int d = 0; d < 32; ++d) o[d] = 0.f;
      for (int j = 0; j < 20; ++j) {
        const float* vrow = ((j < 4) ? (mkv + 2048 + h * 256 + j * 64)
                                     : (qkv + (j - 4) * QS + 1024 + h * 64)) + dbase;
        float pj = p[j] * inv;
        #pragma unroll 8
        for (int d = 0; d < 32; ++d) o[d] += pj * vrow[d];
      }
      float* orow = qkv + m * QS + h * 64 + dbase;  // overwrite q slot
      #pragma unroll
      for (int d = 0; d < 32; ++d) orow[d] = o[d];
    }
    __syncthreads();

    // stage attn-out as f16 A operand
    for (int idx = tid; idx < 8192; idx += 256) {
      int cc = idx >> 4, tt = idx & 15;
      a16[tt * SA + cc] = (_Float16)qkv[tt * QS + cc];
    }
    __syncthreads();

    // GEMM2: [16x512] = out @ wn_out^T  (32 N-tiles, 4 per wave), fused mp_add
    v8f acc2[4];
    for (int j = 0; j < 4; ++j) {
      int n0 = (wv * 4 + j) * 16;
      if (j + 1 < 4)
        __builtin_prefetch(WO + (size_t)(n0 + 16 + mrow) * 512 + kOff, 0, 0);
      v8f acc = {};
      #pragma unroll 4
      for (int kb = 0; kb < 16; ++kb) {
        int k0 = kb * 32 + kOff;
        v16h a = frag_ld(a16 + mrow * SA + k0);
        v16h b = frag_ld(WO + (size_t)(n0 + mrow) * 512 + k0);
        acc = __builtin_amdgcn_wmma_f32_16x16x32_f16(false, a, false, b,
                                                     (short)0, acc, false, false);
      }
      acc2[j] = acc;
    }
    __syncthreads();  // all a16 reads done before overwrite
    for (int j = 0; j < 4; ++j) {
      int n0 = (wv * 4 + j) * 16;
      #pragma unroll
      for (int i = 0; i < 8; ++i) {
        int m = mb + i, n = n0 + mrow;
        float yn = (acc2[j][i] * (1.f - MP_T) + ycur[m * SF + n] * MP_T) * rs;
        ycur[m * SF + n] = yn;
        a16 [m * SA + n] = (_Float16)yn;
      }
    }
    __syncthreads();
  }

  // final proj (gain folded) + outer residual, scatter back to [b,c,t,h,w]
  for (int j = 0; j < 4; ++j) {
    int n0 = (wv * 4 + j) * 16;
    if (j + 1 < 4)
      __builtin_prefetch(WPROJ + (size_t)(n0 + 16 + mrow) * 512 + kOff, 0, 0);
    v8f acc = {};
    #pragma unroll 4
    for (int kb = 0; kb < 16; ++kb) {
      int k0 = kb * 32 + kOff;
      v16h a = frag_ld(a16 + mrow * SA + k0);
      v16h b = frag_ld(WPROJ + (size_t)(n0 + mrow) * 512 + k0);
      acc = __builtin_amdgcn_wmma_f32_16x16x32_f16(false, a, false, b,
                                                   (short)0, acc, false, false);
    }
    #pragma unroll
    for (int i = 0; i < 8; ++i) {
      int m = mb + i, n = n0 + mrow;
      float yv = (acc[i] * (1.f - MP_T) + res0[m * SF + n] * MP_T) * rs;
      out[xbase + (size_t)n * 16384 + (size_t)m * 1024] = yv;
    }
  }
}

extern "C" void kernel_launch(void* const* d_in, const int* in_sizes, int n_in,
                              void* d_out, int out_size, void* d_ws, size_t ws_size,
                              hipStream_t stream) {
  const float* x     = (const float*)d_in[0];
  const float* wqkv  = (const float*)d_in[1];
  const float* wout  = (const float*)d_in[2];
  const float* memkv = (const float*)d_in[3];
  const float* wproj = (const float*)d_in[4];
  const float* gain  = (const float*)d_in[5];
  _Float16* ws16 = (_Float16*)d_ws;
  float*    mkvn = (float*)((char*)d_ws + WS_MKV_BYTE);

  norm_weights<<<576, 256, 0, stream>>>(wqkv, wout, wproj, gain, ws16);
  norm_memkv  <<<16,  256, 0, stream>>>(memkv, mkvn);

  const size_t smem = (size_t)(16 * QS + 2 * 16 * SF + 4096) * 4 + (size_t)(16 * SA) * 2;
  mp_attn_main<<<2048, 256, smem, stream>>>(x, ws16, mkvn, (float*)d_out);
}
